// AugmentedLatentDynamics_50242527428860
// MI455X (gfx1250) — matrix operand order repaired and checked by reading, not compiled, not run on hardware
//
#include <hip/hip_runtime.h>

// MI455X / gfx1250, wave32. f16 WMMA (16x16x32) + f32 accumulate.
// out is (16384, 65) f32: cols 0..63 = dz_dt, col 64 = -divergence.

typedef __attribute__((ext_vector_type(16))) _Float16 v16h;
typedef __attribute__((ext_vector_type(8)))  _Float16 v8h;
typedef __attribute__((ext_vector_type(8)))  float    v8f;

#define ZDIM   64
#define HID    256
#define NBATCH 16384
#define BT     16    // batch rows per workgroup
#define SW     264   // padded LDS row stride (halfs) for 256-wide rows: 528B -> conflict-free b128 lane spread
#define SZ     72    // padded LDS row stride (halfs) for 64-wide rows

// workspace layout in halfs
#define OFF_W1H 0
#define OFF_W2H (16384)
#define OFF_W3H (16384 + 65536)
#define OFF_W4H (16384 + 65536 + 65536)
#define OFF_G1  (163840)
#define OFF_G2  (OFF_G1 + (size_t)NBATCH * HID)
#define OFF_G3  (OFF_G2 + (size_t)NBATCH * HID)

// ---------------- CDNA5 feature guards ----------------
#if __has_builtin(__builtin_amdgcn_global_load_async_to_lds_b128) && \
    __has_builtin(__builtin_amdgcn_s_wait_asynccnt)
#define HAVE_ASYNC_LDS 1
typedef int v4i __attribute__((vector_size(16)));
typedef __attribute__((address_space(1))) v4i glob_v4i;   // "__device__" in clang diags
typedef __attribute__((address_space(3))) v4i lds_v4i;    // "__shared__" in clang diags

__device__ __forceinline__ void async_copy16(const void* gsrc, void* ldst) {
    __builtin_amdgcn_global_load_async_to_lds_b128(
        (glob_v4i*)(v4i*)gsrc, (lds_v4i*)(v4i*)ldst, 0, 0);
}
#else
#define HAVE_ASYNC_LDS 0
#endif

__device__ __forceinline__ float fast_tanh(float x) {
#if __has_builtin(__builtin_amdgcn_tanhf)
    return __builtin_amdgcn_tanhf(x);         // v_tanh_f32 (CDNA5 TRANS op)
#elif __has_builtin(__builtin_amdgcn_tanh_f32)
    return __builtin_amdgcn_tanh_f32(x);
#else
    return tanhf(x);
#endif
}

// ---------------- fragment loaders (ISA 7.12.2 layouts) ----------------
// A 16x32 f16: lane r = lane&15, half = lane>>4.
//   elems e<8 : k = half*8 + e ; elems e>=8: k = 16 + half*8 + (e-8)
__device__ __forceinline__ v16h load_frag_a(const _Float16* row, int half) {
    v8h lo = *(const v8h*)(row + half * 8);
    v8h hi = *(const v8h*)(row + 16 + half * 8);
    return __builtin_shufflevector(lo, hi, 0,1,2,3,4,5,6,7,8,9,10,11,12,13,14,15);
}
// B 32x16 f16: lane n = lane&15; same k order as A: e<8: k = half*8 + e ; e>=8: k = 16 + half*8 + (e-8).
__device__ __forceinline__ v16h load_frag_b(const _Float16* row, int half) {
    v8h lo = *(const v8h*)(row + half * 8);
    v8h hi = *(const v8h*)(row + 16 + half * 8);
    return __builtin_shufflevector(lo, hi, 0,1,2,3,4,5,6,7,8,9,10,11,12,13,14,15);
}

__device__ __forceinline__ v8f wmma_step(v16h a, v16h b, v8f c) {
    return __builtin_amdgcn_wmma_f32_16x16x32_f16(false, a, false, b, (short)0, c, false, false);
}

// Two output tiles sharing the A fragment (forward layers: one batch A-tile, two N-tiles).
template <int KT>
__device__ __forceinline__ void wmma_dual_sharedA(const _Float16* Arow,
                                                  const _Float16* Brow0,
                                                  const _Float16* Brow1,
                                                  int half, v8f& acc0, v8f& acc1) {
#pragma unroll
    for (int kt = 0; kt < KT; ++kt) {
        v16h a  = load_frag_a(Arow  + kt * 32, half);
        v16h b0 = load_frag_b(Brow0 + kt * 32, half);
        v16h b1 = load_frag_b(Brow1 + kt * 32, half);
        acc0 = wmma_step(a, b0, acc0);
        acc1 = wmma_step(a, b1, acc1);
    }
}
// Two output tiles sharing the B fragment (divergence GEMMs: two M-tiles, one dir-tile).
template <int KT>
__device__ __forceinline__ void wmma_dual_sharedB(const _Float16* Arow0,
                                                  const _Float16* Arow1,
                                                  const _Float16* Brow,
                                                  int half, v8f& acc0, v8f& acc1) {
#pragma unroll
    for (int kt = 0; kt < KT; ++kt) {
        v16h b  = load_frag_b(Brow  + kt * 32, half);
        v16h a0 = load_frag_a(Arow0 + kt * 32, half);
        v16h a1 = load_frag_a(Arow1 + kt * 32, half);
        acc0 = wmma_step(a0, b, acc0);
        acc1 = wmma_step(a1, b, acc1);
    }
}
// Single tile (last forward layer).
template <int KT>
__device__ __forceinline__ v8f wmma_single(const _Float16* Arow, const _Float16* Brow, int half) {
    v8f acc = {};
#pragma unroll
    for (int kt = 0; kt < KT; ++kt) {
        v16h a = load_frag_a(Arow + kt * 32, half);
        v16h b = load_frag_b(Brow + kt * 32, half);
        acc = wmma_step(a, b, acc);
    }
    return acc;
}

// Stage packed 256x256 f16 W2/W3 into padded LDS rows (33 uint4 per row, 32 used).
__device__ __forceinline__ void stage_weights(const _Float16* wsh, _Float16* W2s,
                                              _Float16* W3s, int tid) {
    const uint4* s2 = (const uint4*)(wsh + OFF_W2H);
    const uint4* s3 = (const uint4*)(wsh + OFF_W3H);
    uint4* d2 = (uint4*)W2s;
    uint4* d3 = (uint4*)W3s;
#if HAVE_ASYNC_LDS
    for (int i = tid; i < 8192; i += 256) {
        int row = i >> 5, c = i & 31;
        async_copy16(s2 + i, d2 + row * 33 + c);
        async_copy16(s3 + i, d3 + row * 33 + c);
    }
    __builtin_amdgcn_s_wait_asynccnt(0);
#else
    for (int i = tid; i < 8192; i += 256) {
        int row = i >> 5, c = i & 31;
        d2[row * 33 + c] = s2[i];
        d3[row * 33 + c] = s3[i];
    }
#endif
}

// ---------------- kernel 0: f32 -> f16 weight conversion ----------------
__global__ void cvt_kernel(const float* __restrict__ src, _Float16* __restrict__ dst, int n) {
    int i = blockIdx.x * blockDim.x + threadIdx.x;
    if (i < n) dst[i] = (_Float16)src[i];
}

// ---------------- kernel 1: forward pass ----------------
__global__ void fwd_kernel(const float* __restrict__ states,
                           const float* __restrict__ b1f, const float* __restrict__ b2f,
                           const float* __restrict__ b3f, const float* __restrict__ b4f,
                           _Float16* __restrict__ wsh,
                           float* __restrict__ out) {
    extern __shared__ char smem[];
    _Float16* W2s = (_Float16*)smem;         // 256*SW
    _Float16* W3s = W2s + 256 * SW;          // 256*SW
    _Float16* Zt  = W3s + 256 * SW;          // BT*SZ
    _Float16* Ha  = Zt + BT * SZ;            // BT*SW
    _Float16* Hb  = Ha + BT * SW;            // BT*SW

    const int tid  = threadIdx.x;
    const int lane = tid & 31;
    const int wv   = tid >> 5;
    const int half = lane >> 4;
    const int nl   = lane & 15;
    const size_t b0 = (size_t)blockIdx.x * BT;

    stage_weights(wsh, W2s, W3s, tid);
    for (int i = tid; i < BT * ZDIM; i += 256) {
        int m = i >> 6, k = i & 63;
        Zt[m * SZ + k] = (_Float16)states[(b0 + m) * (ZDIM + 1) + k];
    }
    __syncthreads();

    const _Float16* w1h = wsh + OFF_W1H;
    const _Float16* w4h = wsh + OFF_W4H;
    _Float16* g1 = wsh + OFF_G1 + b0 * HID;
    _Float16* g2 = wsh + OFF_G2 + b0 * HID;
    _Float16* g3 = wsh + OFF_G3 + b0 * HID;

    const int nt0 = wv * 2, nt1 = wv * 2 + 1;
    const int n0 = nt0 * 16 + nl, n1 = nt1 * 16 + nl;

    // ---- layer 1: h1 = tanh(Z @ W1^T + b1) ----
    {
        v8f a0 = {}, a1 = {};
        wmma_dual_sharedA<2>(Zt + nl * SZ, w1h + n0 * ZDIM, w1h + n1 * ZDIM, half, a0, a1);
        float bb0 = b1f[n0], bb1 = b1f[n1];
#pragma unroll
        for (int v = 0; v < 8; ++v) {
            int m = v + 8 * half;
            float h0 = fast_tanh(a0[v] + bb0);
            float h1 = fast_tanh(a1[v] + bb1);
            Ha[m * SW + n0] = (_Float16)h0;
            Ha[m * SW + n1] = (_Float16)h1;
            g1[(size_t)m * HID + n0] = (_Float16)(1.0f - h0 * h0);
            g1[(size_t)m * HID + n1] = (_Float16)(1.0f - h1 * h1);
        }
    }
    __syncthreads();

    // ---- layer 2: h2 = tanh(h1 @ W2^T + b2) ----
    {
        v8f a0 = {}, a1 = {};
        wmma_dual_sharedA<8>(Ha + nl * SW, W2s + n0 * SW, W2s + n1 * SW, half, a0, a1);
        float bb0 = b2f[n0], bb1 = b2f[n1];
#pragma unroll
        for (int v = 0; v < 8; ++v) {
            int m = v + 8 * half;
            float h0 = fast_tanh(a0[v] + bb0);
            float h1 = fast_tanh(a1[v] + bb1);
            Hb[m * SW + n0] = (_Float16)h0;
            Hb[m * SW + n1] = (_Float16)h1;
            g2[(size_t)m * HID + n0] = (_Float16)(1.0f - h0 * h0);
            g2[(size_t)m * HID + n1] = (_Float16)(1.0f - h1 * h1);
        }
    }
    __syncthreads();

    // ---- layer 3: h3 = tanh(h2 @ W3^T + b3) ----
    {
        v8f a0 = {}, a1 = {};
        wmma_dual_sharedA<8>(Hb + nl * SW, W3s + n0 * SW, W3s + n1 * SW, half, a0, a1);
        float bb0 = b3f[n0], bb1 = b3f[n1];
#pragma unroll
        for (int v = 0; v < 8; ++v) {
            int m = v + 8 * half;
            float h0 = fast_tanh(a0[v] + bb0);
            float h1 = fast_tanh(a1[v] + bb1);
            Ha[m * SW + n0] = (_Float16)h0;
            Ha[m * SW + n1] = (_Float16)h1;
            g3[(size_t)m * HID + n0] = (_Float16)(1.0f - h0 * h0);
            g3[(size_t)m * HID + n1] = (_Float16)(1.0f - h1 * h1);
        }
    }
    __syncthreads();

    // ---- layer 4: dz = h3 @ W4^T + b4 (4 N-tiles -> waves 0..3) ----
    if (wv < 4) {
        int n = wv * 16 + nl;
        v8f acc = wmma_single<8>(Ha + nl * SW, w4h + n * HID, half);
        float bb = b4f[n];
#pragma unroll
        for (int v = 0; v < 8; ++v) {
            int m = v + 8 * half;
            out[(b0 + m) * (ZDIM + 1) + n] = acc[v] + bb;
        }
    }
}

// ---------------- kernel 2: divergence (Jacobian trace) ----------------
// div_b = sum_{q,i} (W3 V)[q,i] * g3[q] * W4[i,q],  V = diag(g2) (W2 diag(g1) W1)
__global__ void div_kernel(_Float16* __restrict__ wsh, float* __restrict__ out) {
    extern __shared__ char smem[];
    _Float16* W2s = (_Float16*)smem;          // 256*SW
    _Float16* W3s = W2s + 256 * SW;           // 256*SW
    _Float16* Ut  = W3s + 256 * SW;           // 16*SW (dir-major)
    _Float16* Vt  = Ut + 16 * SW;             // 16*SW (dir-major)
    float* g1L = (float*)(Vt + 16 * SW);      // 256
    float* g2L = g1L + HID;
    float* g3L = g2L + HID;
    float* divsh = g3L + HID;                 // 1 (+pad)

    const int tid  = threadIdx.x;
    const int lane = tid & 31;
    const int wv   = tid >> 5;
    const int half = lane >> 4;
    const int nl   = lane & 15;
    const size_t b0 = (size_t)blockIdx.x * BT;

    stage_weights(wsh, W2s, W3s, tid);
    const _Float16* w1h = wsh + OFF_W1H;
    const _Float16* w4h = wsh + OFF_W4H;
    __builtin_prefetch(w4h + tid * 64, 0, 1);    // warm L2/WGP$ for W4 rows
    __syncthreads();

    const int mt0 = wv * 2, mt1 = wv * 2 + 1;

    for (int bb = 0; bb < BT; ++bb) {
        const size_t b = b0 + bb;
        g1L[tid] = (float)wsh[OFF_G1 + b * HID + tid];
        g2L[tid] = (float)wsh[OFF_G2 + b * HID + tid];
        g3L[tid] = (float)wsh[OFF_G3 + b * HID + tid];
        if (tid == 0) divsh[0] = 0.0f;
        __syncthreads();

        float psum = 0.0f;
        for (int c = 0; c < 4; ++c) {   // 4 chunks of 16 tangent directions
            // build U chunk: Ut[i][p] = g1[p] * W1[p][c*16+i]
            {
                const int p = tid;
                v8h wa = *(const v8h*)(w1h + p * ZDIM + c * 16);
                v8h wb = *(const v8h*)(w1h + p * ZDIM + c * 16 + 8);
                float g = g1L[p];
#pragma unroll
                for (int i = 0; i < 8; ++i) {
                    Ut[i * SW + p]       = (_Float16)(g * (float)wa[i]);
                    Ut[(i + 8) * SW + p] = (_Float16)(g * (float)wb[i]);
                }
            }
            __syncthreads();

            // GEMM 1: V = diag(g2) * (W2 @ U)   M=256 K=256 N=16
            {
                v8f a0 = {}, a1 = {};
                wmma_dual_sharedB<8>(W2s + (mt0 * 16 + nl) * SW,
                                     W2s + (mt1 * 16 + nl) * SW,
                                     Ut + nl * SW, half, a0, a1);
#pragma unroll
                for (int v = 0; v < 8; ++v) {
                    int q0 = mt0 * 16 + v + 8 * half;
                    int q1 = mt1 * 16 + v + 8 * half;
                    Vt[nl * SW + q0] = (_Float16)(a0[v] * g2L[q0]);
                    Vt[nl * SW + q1] = (_Float16)(a1[v] * g2L[q1]);
                }
            }
            __syncthreads();

            // GEMM 2: P = W3 @ V, fold g3[q]*W4[i][q] into the reduction
            {
                v8f a0 = {}, a1 = {};
                wmma_dual_sharedB<8>(W3s + (mt0 * 16 + nl) * SW,
                                     W3s + (mt1 * 16 + nl) * SW,
                                     Vt + nl * SW, half, a0, a1);
                v8h w40 = *(const v8h*)(w4h + (size_t)(c * 16 + nl) * HID + mt0 * 16 + half * 8);
                v8h w41 = *(const v8h*)(w4h + (size_t)(c * 16 + nl) * HID + mt1 * 16 + half * 8);
#pragma unroll
                for (int v = 0; v < 8; ++v) {
                    int q0 = mt0 * 16 + v + 8 * half;
                    int q1 = mt1 * 16 + v + 8 * half;
                    psum += a0[v] * g3L[q0] * (float)w40[v];
                    psum += a1[v] * g3L[q1] * (float)w41[v];
                }
            }
            __syncthreads();   // Ut/Vt reuse next chunk
        }
        atomicAdd(&divsh[0], psum);   // ds_add_f32
        __syncthreads();
        if (tid == 0) out[b * (ZDIM + 1) + ZDIM] = -divsh[0];
        __syncthreads();
    }
}

extern "C" void kernel_launch(void* const* d_in, const int* in_sizes, int n_in,
                              void* d_out, int out_size, void* d_ws, size_t ws_size,
                              hipStream_t stream) {
    // inputs: 0 t, 1 states, 2 W1, 3 b1, 4 W2, 5 b2, 6 W3, 7 b3, 8 W4, 9 b4
    const float* states = (const float*)d_in[1];
    const float* W1 = (const float*)d_in[2];
    const float* b1 = (const float*)d_in[3];
    const float* W2 = (const float*)d_in[4];
    const float* b2 = (const float*)d_in[5];
    const float* W3 = (const float*)d_in[6];
    const float* b3 = (const float*)d_in[7];
    const float* W4 = (const float*)d_in[8];
    const float* b4 = (const float*)d_in[9];
    float* out = (float*)d_out;
    _Float16* wsh = (_Float16*)d_ws;

    cvt_kernel<<<16384 / 256, 256, 0, stream>>>(W1, wsh + OFF_W1H, 16384);
    cvt_kernel<<<65536 / 256, 256, 0, stream>>>(W2, wsh + OFF_W2H, 65536);
    cvt_kernel<<<65536 / 256, 256, 0, stream>>>(W3, wsh + OFF_W3H, 65536);
    cvt_kernel<<<16384 / 256, 256, 0, stream>>>(W4, wsh + OFF_W4H, 16384);

    size_t lds1 = (size_t)(2 * 256 * SW + BT * SZ + 2 * BT * SW) * sizeof(_Float16);
    fwd_kernel<<<NBATCH / BT, 256, lds1, stream>>>(states, b1, b2, b3, b4, wsh, out);

    size_t lds2 = (size_t)(2 * 256 * SW + 2 * 16 * SW) * sizeof(_Float16)
                + (size_t)(3 * HID + 8) * sizeof(float);
    div_kernel<<<NBATCH / BT, 256, lds2, stream>>>(wsh, out);
}
